// Q2LTCModel_26895085207873
// MI455X (gfx1250) — compile-verified, matrix operand order checked by reading, not run on hardware
//
#include <hip/hip_runtime.h>
#include <hip/hip_bf16.h>

// ---------------------------------------------------------------------------
// Q2LTC forward on gfx1250 (MI455X). Compile-only target.
// Dense math: v_wmma_f32_16x16x32_bf16 (bf16 in, f32 accum).
// GEMM staging: double-buffered LDS; GLOBAL_LOAD_ASYNC_TO_LDS_B128 when the
// toolchain exposes the builtin (ASYNCcnt-tracked DMA, no VGPR round trip),
// otherwise synchronous global->VGPR->LDS fallback.
// Weights pre-converted fp32->bf16 once per launch (halves L2/LDS traffic).
// ---------------------------------------------------------------------------

typedef __bf16 bf16;
typedef __attribute__((ext_vector_type(16))) __bf16 v16bf;
typedef __attribute__((ext_vector_type(8)))  float  v8f;
typedef int vint4 __attribute__((vector_size(16)));   // matches builtin param type

#define B_    2
#define T_    2048
#define D_    768
#define V_    1024
#define H_    12
#define HKV_  4
#define HD_   64
#define KVD_  (HKV_*HD_)   // 256
#define DFF_  2304
#define NG_   3
#define M_    (B_*T_)      // 4096
#define EPS_  1e-6f
#define CFC_BLOCKS 48

#if defined(__HIP_DEVICE_COMPILE__) && __has_builtin(__builtin_amdgcn_global_load_async_to_lds_b128)
#define USE_ASYNC_LDS 1
#else
#define USE_ASYNC_LDS 0
#endif

#if USE_ASYNC_LDS
__device__ __forceinline__ void async_g2l_16B(const void* g, void* l) {
  // GLOBAL_LOAD_ASYNC_TO_LDS_B128: per-lane 16B global -> LDS, ASYNCcnt.
  __builtin_amdgcn_global_load_async_to_lds_b128(
      (__attribute__((address_space(1))) vint4*)g,
      (__attribute__((address_space(3))) vint4*)l, /*offset=*/0, /*cpol=*/0);
}
__device__ __forceinline__ void wait_async0() {
#if __has_builtin(__builtin_amdgcn_s_wait_asynccnt)
  __builtin_amdgcn_s_wait_asynccnt(0);
#else
  asm volatile("s_wait_asynccnt 0x0" ::: "memory");
#endif
}
#endif

// ---------------------------------------------------------------------------
// RMSNorm: one block (256 threads) per row of D_=768.
// ---------------------------------------------------------------------------
__global__ __launch_bounds__(256)
void k_rmsnorm(const float* __restrict__ x, const float* __restrict__ w,
               float* __restrict__ out) {
  const int row = blockIdx.x;
  const float* xr = x + (size_t)row * D_;
  __shared__ float red[256];
  float s = 0.f;
  for (int i = threadIdx.x; i < D_; i += 256) { float v = xr[i]; s += v * v; }
  red[threadIdx.x] = s;
  __syncthreads();
  for (int st = 128; st > 0; st >>= 1) {
    if (threadIdx.x < st) red[threadIdx.x] += red[threadIdx.x + st];
    __syncthreads();
  }
  const float inv = rsqrtf(red[0] * (1.0f / D_) + EPS_);
  for (int i = threadIdx.x; i < D_; i += 256)
    out[(size_t)row * D_ + i] = xr[i] * inv * w[i];
}

__global__ __launch_bounds__(256)
void k_embed_rmsnorm(const int* __restrict__ tokens, const float* __restrict__ embed,
                     const float* __restrict__ w, float* __restrict__ out) {
  const int row = blockIdx.x;
  const int tok = tokens[row];
  const float* xr = embed + (size_t)tok * D_;
  __shared__ float red[256];
  float s = 0.f;
  for (int i = threadIdx.x; i < D_; i += 256) { float v = xr[i]; s += v * v; }
  red[threadIdx.x] = s;
  __syncthreads();
  for (int st = 128; st > 0; st >>= 1) {
    if (threadIdx.x < st) red[threadIdx.x] += red[threadIdx.x + st];
    __syncthreads();
  }
  const float inv = rsqrtf(red[0] * (1.0f / D_) + EPS_);
  for (int i = threadIdx.x; i < D_; i += 256)
    out[(size_t)row * D_ + i] = xr[i] * inv * w[i];
}

// fp32 -> bf16 weight conversion (once per launch; weights are static).
__global__ __launch_bounds__(256)
void k_f32_to_bf16(const float* __restrict__ s, bf16* __restrict__ d, int n) {
  const int i = blockIdx.x * 256 + threadIdx.x;
  if (i < n) d[i] = (bf16)s[i];
}

// ---------------------------------------------------------------------------
// GEMM: C[M,N] = A[M,K] @ W[N,K]^T.  A fp32 global, W bf16 (pre-converted).
// Block tile 64(M) x 128(N), 8 waves, each wave 32x32 (2x2 WMMA C tiles).
// Double-buffered LDS pipeline: issue tile k+1 (async when available) while
// computing tile k; s_wait_asynccnt 0 + barrier at the top of each iteration.
// EPI: 0 = plain, 1 = +R residual, 2 = silu(R) * result.
// ---------------------------------------------------------------------------
template<int EPI>
__global__ __launch_bounds__(256)
void k_gemm_bf16(const float* __restrict__ A, const bf16* __restrict__ W,
                 const float* __restrict__ R, float* __restrict__ C,
                 int N, int K) {
  __shared__ float sA[2][64 * 32];   // A tile, fp32 (converted at gather)
  __shared__ bf16  sB[2][128 * 32];  // W tile, bf16
  const int m0   = blockIdx.y * 64;
  const int n0   = blockIdx.x * 128;
  const int tid  = threadIdx.x;
  const int lane = tid & 31;
  const int wave = tid >> 5;
  const int wm   = wave >> 2;   // 0..1
  const int wn   = wave & 3;    // 0..3
  const int laneN = lane & 15;
  const int kbase = (lane & 16) ? 8 : 0;

  v8f acc[2][2] = {};

  auto stage = [&](int k0, int buf) {
#if USE_ASYNC_LDS
    // A: 64 rows x 32 f32 = 512 x 16B chunks (2/thread).
    for (int i = tid; i < 512; i += 256) {
      const int r = i >> 3, c = (i & 7) * 4;
      async_g2l_16B(&A[(size_t)(m0 + r) * K + k0 + c], &sA[buf][r * 32 + c]);
    }
    // B: 128 rows x 32 bf16 = 512 x 16B chunks (2/thread).
    for (int i = tid; i < 512; i += 256) {
      const int r = i >> 2, c = (i & 3) * 8;
      async_g2l_16B(&W[(size_t)(n0 + r) * K + k0 + c], &sB[buf][r * 32 + c]);
    }
#else
    for (int i = tid; i < 64 * 32; i += 256) {
      const int r = i >> 5, c = i & 31;
      sA[buf][i] = A[(size_t)(m0 + r) * K + k0 + c];
    }
    for (int i = tid; i < 128 * 32; i += 256) {
      const int r = i >> 5, c = i & 31;
      sB[buf][i] = W[(size_t)(n0 + r) * K + k0 + c];
    }
    if (k0 + 32 < K) {   // global_prefetch_b8 of the tile after next
      __builtin_prefetch(&A[(size_t)(m0 + (tid >> 3)) * K + k0 + 32 + (tid & 7) * 4], 0, 1);
      __builtin_prefetch(&W[(size_t)(n0 + (tid >> 1)) * K + k0 + 32 + (tid & 1) * 16], 0, 1);
    }
#endif
  };

  const int nk = K >> 5;
  stage(0, 0);
  int cur = 0;
  for (int kt = 0; kt < nk; ++kt) {
#if USE_ASYNC_LDS
    wait_async0();                 // this wave's in-flight tile is in LDS
#endif
    __syncthreads();               // every wave's staging of buf[cur] visible
    if (kt + 1 < nk) stage((kt + 1) << 5, cur ^ 1);

    v16bf fa[2], fb[2];
#pragma unroll
    for (int t = 0; t < 2; ++t) {
      const int ar = wm * 32 + t * 16 + laneN;   // A row (M)
      const int br = wn * 32 + t * 16 + laneN;   // W row (N)
#pragma unroll
      for (int e = 0; e < 16; ++e) {
        const int kk = kbase + (e < 8 ? e : e + 8);
        fa[t][e] = (bf16)sA[cur][ar * 32 + kk];
        fb[t][e] = sB[cur][br * 32 + kk];
      }
    }
#pragma unroll
    for (int ti = 0; ti < 2; ++ti)
#pragma unroll
      for (int tj = 0; tj < 2; ++tj)
        acc[ti][tj] = __builtin_amdgcn_wmma_f32_16x16x32_bf16(
            false, fa[ti], false, fb[tj], (short)0, acc[ti][tj], false, false);
    cur ^= 1;
  }

  // C-tile layout (ISA 7.12.2): VGPR r, lane<16 -> M=r, lane>=16 -> M=r+8.
  const int rOff = (lane & 16) ? 8 : 0;
#pragma unroll
  for (int ti = 0; ti < 2; ++ti)
#pragma unroll
    for (int tj = 0; tj < 2; ++tj)
#pragma unroll
      for (int r = 0; r < 8; ++r) {
        const int row = m0 + wm * 32 + ti * 16 + rOff + r;
        const int col = n0 + wn * 32 + tj * 16 + laneN;
        const size_t idx = (size_t)row * N + col;
        float v = acc[ti][tj][r];
        if (EPI == 1) v += R[idx];
        if (EPI == 2) { float gv = R[idx]; v *= gv / (1.f + __expf(-gv)); }
        C[idx] = v;
      }
}

// ---------------------------------------------------------------------------
// Causal flash attention. One wave per (16-query tile, head, batch).
// ---------------------------------------------------------------------------
__global__ __launch_bounds__(32)
void k_attn(const float* __restrict__ q, const float* __restrict__ k,
            const float* __restrict__ v, float* __restrict__ o) {
  __shared__ bf16 sK[32 * 64];
  __shared__ bf16 sV[32 * 64];
  __shared__ bf16 sP[16 * 32];
  const int lane  = threadIdx.x;
  const int qi0   = blockIdx.x * 16;
  const int h     = blockIdx.y;
  const int b     = blockIdx.z;
  const int hk    = h / NG_;
  const int laneN = lane & 15;
  const int kbase = (lane & 16) ? 8 : 0;
  const int rOff  = (lane & 16) ? 8 : 0;

  v16bf fq[2];
#pragma unroll
  for (int hh = 0; hh < 2; ++hh)
#pragma unroll
    for (int e = 0; e < 16; ++e) {
      const int kk = kbase + (e < 8 ? e : e + 8);
      fq[hh][e] = (bf16)(q[(size_t)(b * T_ + qi0 + laneN) * D_ + h * HD_ + hh * 32 + kk] * 0.125f);
    }

  v8f ao[4] = {};
  float rowM[8], rowL[8];
#pragma unroll
  for (int r = 0; r < 8; ++r) { rowM[r] = -3.0e38f; rowL[r] = 0.f; }

  const int nkb = (qi0 >> 5) + 1;
  for (int kb = 0; kb < nkb; ++kb) {
    const int t0 = kb * 32;
    for (int i = lane; i < 32 * 64; i += 32) {
      const int kr = i >> 6, d = i & 63;
      const size_t base = (size_t)(b * T_ + t0 + kr) * KVD_ + hk * HD_ + d;
      sK[i] = (bf16)k[base];
      sV[i] = (bf16)v[base];
    }
    __syncthreads();

    v8f sc[2] = {};
#pragma unroll
    for (int sn = 0; sn < 2; ++sn)
#pragma unroll
      for (int hh = 0; hh < 2; ++hh) {
        v16bf fk;
#pragma unroll
        for (int e = 0; e < 16; ++e) {
          const int kk = kbase + (e < 8 ? e : e + 8);
          fk[e] = sK[(sn * 16 + laneN) * 64 + hh * 32 + kk];
        }
        sc[sn] = __builtin_amdgcn_wmma_f32_16x16x32_bf16(
            false, fq[hh], false, fk, (short)0, sc[sn], false, false);
      }

#pragma unroll
    for (int r = 0; r < 8; ++r) {
      const int row = qi0 + rOff + r;
      float s0 = sc[0][r], s1 = sc[1][r];
      const int key0 = t0 + laneN;
      if (key0 > row)      s0 = -1e30f;
      if (key0 + 16 > row) s1 = -1e30f;
      float cand = fmaxf(s0, s1);
#pragma unroll
      for (int off = 1; off < 16; off <<= 1) cand = fmaxf(cand, __shfl_xor(cand, off, 32));
      const float mNew = fmaxf(rowM[r], cand);
      const float corr = __expf(rowM[r] - mNew);
      const float p0 = __expf(s0 - mNew), p1 = __expf(s1 - mNew);
      float psum = p0 + p1;
#pragma unroll
      for (int off = 1; off < 16; off <<= 1) psum += __shfl_xor(psum, off, 32);
      rowL[r] = rowL[r] * corr + psum;
      rowM[r] = mNew;
#pragma unroll
      for (int c = 0; c < 4; ++c) ao[c][r] *= corr;
      sP[(rOff + r) * 32 + laneN]      = (bf16)p0;
      sP[(rOff + r) * 32 + 16 + laneN] = (bf16)p1;
    }
    __syncthreads();

    v16bf fp;
#pragma unroll
    for (int e = 0; e < 16; ++e) {
      const int kk = kbase + (e < 8 ? e : e + 8);
      fp[e] = sP[laneN * 32 + kk];
    }
#pragma unroll
    for (int c = 0; c < 4; ++c) {
      v16bf fv;
#pragma unroll
      for (int e = 0; e < 16; ++e) {
        const int kk = kbase + (e < 8 ? e : e + 8);
        fv[e] = sV[kk * 64 + c * 16 + laneN];
      }
      ao[c] = __builtin_amdgcn_wmma_f32_16x16x32_bf16(
          false, fp, false, fv, (short)0, ao[c], false, false);
    }
    __syncthreads();
  }

#pragma unroll
  for (int r = 0; r < 8; ++r) {
    const float inv = 1.f / rowL[r];
    const int row = qi0 + rOff + r;
#pragma unroll
    for (int c = 0; c < 4; ++c)
      o[(size_t)(b * T_ + row) * D_ + h * HD_ + c * 16 + laneN] = ao[c][r] * inv;
  }
}

// ---------------------------------------------------------------------------
// CfC closed-form LTC scan: persistent grid-resident kernel with a
// device-wide generation-counter barrier per timestep.
// ---------------------------------------------------------------------------
__global__ __launch_bounds__(256)
void k_cfc_scan(const float* __restrict__ xn, const float* __restrict__ Wa1,
                const float* __restrict__ Wa2, const float* __restrict__ logdt,
                float* __restrict__ hbuf, float* __restrict__ y,
                unsigned* __restrict__ ctr) {
  __shared__ float xh[2 * D_];
  __shared__ float red1[256], red2[256];
  const int tid  = threadIdx.x;
  const int pair = tid >> 3, part = tid & 7;
  const int p = blockIdx.x * 32 + pair;          // global (b,d)
  const int b = p / D_, d = p - b * D_;          // same b per block
  const float dt = __expf(logdt[d]);
  const int j0 = part * 192;
  const float* wa1 = Wa1 + (size_t)d * (2 * D_);
  const float* wa2 = Wa2 + (size_t)d * (2 * D_);

  for (int t = 0; t < T_; ++t) {
    for (int i = tid; i < 2 * D_; i += 256)
      xh[i] = (i < D_) ? xn[(size_t)(b * T_ + t) * D_ + i] : hbuf[b * D_ + (i - D_)];
    __syncthreads();

    float s1 = 0.f, s2 = 0.f;
#pragma unroll 8
    for (int j = j0; j < j0 + 192; ++j) {
      const float xv = xh[j];
      s1 = fmaf(wa1[j], xv, s1);
      s2 = fmaf(wa2[j], xv, s2);
    }
    red1[tid] = s1; red2[tid] = s2;
    __syncthreads();

    if (part == 0) {
      float a1 = 0.f, a2 = 0.f;
#pragma unroll
      for (int qq = 0; qq < 8; ++qq) { a1 += red1[tid + qq]; a2 += red2[tid + qq]; }
      const float sp    = (a1 > 30.f) ? a1 : log1pf(__expf(a1));
      const float decay = __expf(-sp * dt);
      const float hOld  = xh[D_ + d];
      const float hNew  = decay * hOld + (a2 / (sp + EPS_)) * (1.f - decay);
      hbuf[b * D_ + d] = hNew;
      y[(size_t)(b * T_ + t) * D_ + d] = hNew;
    }
    __threadfence();
    __syncthreads();

    if (tid == 0) {
      atomicAdd(ctr, 1u);
      const unsigned target = (unsigned)gridDim.x * (unsigned)(t + 1);
      while (atomicAdd(ctr, 0u) < target) { __builtin_amdgcn_s_sleep(1); }
    }
    __syncthreads();
    __threadfence();
  }
}

__global__ void k_zero(float* __restrict__ pz, int n) {
  const int i = blockIdx.x * 256 + threadIdx.x;
  if (i < n) pz[i] = 0.f;
}

// ---------------------------------------------------------------------------
// Host orchestration. Workspace (~170 MB): fp32 activation buffers + bf16
// weight mirrors + CfC state.
// ---------------------------------------------------------------------------
extern "C" void kernel_launch(void* const* d_in, const int* in_sizes, int n_in,
                              void* d_out, int out_size, void* d_ws, size_t ws_size,
                              hipStream_t stream) {
  (void)in_sizes; (void)n_in; (void)out_size; (void)ws_size;
  const int*   tokens    = (const int*)d_in[0];
  const float* embed     = (const float*)d_in[1];
  const float* emb_norm  = (const float*)d_in[2];
  const float* attn_norm = (const float*)d_in[3];
  const float* Wq  = (const float*)d_in[4];
  const float* Wk  = (const float*)d_in[5];
  const float* Wv  = (const float*)d_in[6];
  const float* Wo  = (const float*)d_in[7];
  const float* mlp_norm = (const float*)d_in[8];
  const float* Wu  = (const float*)d_in[9];
  const float* Wg  = (const float*)d_in[10];
  const float* Wd  = (const float*)d_in[11];
  const float* cnorm  = (const float*)d_in[12];
  const float* Wa1    = (const float*)d_in[13];
  const float* Wa2    = (const float*)d_in[14];
  const float* Wout   = (const float*)d_in[15];
  const float* clogdt = (const float*)d_in[16];
  const float* fnorm  = (const float*)d_in[17];

  char* wsp = (char*)d_ws;
  size_t off = 0;
  auto carve = [&](size_t bytes) -> void* {
    void* pp = (void*)(wsp + off);
    off = (off + bytes + 255) & ~(size_t)255;
    return pp;
  };
  float* xbuf = (float*)carve((size_t)M_ * D_ * 4);
  float* nbuf = (float*)carve((size_t)M_ * D_ * 4);
  float* qbuf = (float*)carve((size_t)M_ * D_ * 4);
  float* abuf = (float*)carve((size_t)M_ * D_ * 4);
  float* kbuf = (float*)carve((size_t)M_ * KVD_ * 4);
  float* vbuf = (float*)carve((size_t)M_ * KVD_ * 4);
  float* gbuf = (float*)carve((size_t)M_ * DFF_ * 4);
  float* hbuf = (float*)carve((size_t)(B_ * D_ + 64) * 4);
  unsigned* ctr = (unsigned*)(hbuf + B_ * D_);
  // bf16 weight mirrors
  bf16* bWq   = (bf16*)carve((size_t)4 * D_ * D_ * 2);
  bf16* bWk   = (bf16*)carve((size_t)4 * KVD_ * D_ * 2);
  bf16* bWv   = (bf16*)carve((size_t)4 * KVD_ * D_ * 2);
  bf16* bWo   = (bf16*)carve((size_t)4 * D_ * D_ * 2);
  bf16* bWu   = (bf16*)carve((size_t)4 * DFF_ * D_ * 2);
  bf16* bWg   = (bf16*)carve((size_t)4 * DFF_ * D_ * 2);
  bf16* bWd   = (bf16*)carve((size_t)4 * D_ * DFF_ * 2);
  bf16* bWout = (bf16*)carve((size_t)12 * D_ * D_ * 2);
  bf16* bEmb  = (bf16*)carve((size_t)V_ * D_ * 2);

  auto cvt = [&](const float* s, bf16* dd, size_t n) {
    k_f32_to_bf16<<<(unsigned)((n + 255) / 256), 256, 0, stream>>>(s, dd, (int)n);
  };
  cvt(Wq,  bWq,  (size_t)4 * D_ * D_);
  cvt(Wk,  bWk,  (size_t)4 * KVD_ * D_);
  cvt(Wv,  bWv,  (size_t)4 * KVD_ * D_);
  cvt(Wo,  bWo,  (size_t)4 * D_ * D_);
  cvt(Wu,  bWu,  (size_t)4 * DFF_ * D_);
  cvt(Wg,  bWg,  (size_t)4 * DFF_ * D_);
  cvt(Wd,  bWd,  (size_t)4 * D_ * DFF_);
  cvt(Wout, bWout, (size_t)12 * D_ * D_);
  cvt(embed, bEmb, (size_t)V_ * D_);

  const dim3 gD  (D_   / 128, M_ / 64);
  const dim3 gKV (KVD_ / 128, M_ / 64);
  const dim3 gFF (DFF_ / 128, M_ / 64);
  const dim3 gV  (V_   / 128, M_ / 64);

  k_embed_rmsnorm<<<M_, 256, 0, stream>>>(tokens, embed, emb_norm, xbuf);

  for (int g = 0; g < 4; ++g) {
    const bf16* wq = bWq + (size_t)g * D_ * D_;
    const bf16* wk = bWk + (size_t)g * KVD_ * D_;
    const bf16* wv = bWv + (size_t)g * KVD_ * D_;
    const bf16* wo = bWo + (size_t)g * D_ * D_;
    const bf16* wu = bWu + (size_t)g * DFF_ * D_;
    const bf16* wg = bWg + (size_t)g * DFF_ * D_;
    const bf16* wd = bWd + (size_t)g * D_ * DFF_;

    k_rmsnorm<<<M_, 256, 0, stream>>>(xbuf, attn_norm + g * D_, nbuf);
    k_gemm_bf16<0><<<gD,  256, 0, stream>>>(nbuf, wq, nullptr, qbuf, D_,   D_);
    k_gemm_bf16<0><<<gKV, 256, 0, stream>>>(nbuf, wk, nullptr, kbuf, KVD_, D_);
    k_gemm_bf16<0><<<gKV, 256, 0, stream>>>(nbuf, wv, nullptr, vbuf, KVD_, D_);
    k_attn<<<dim3(T_ / 16, H_, B_), 32, 0, stream>>>(qbuf, kbuf, vbuf, abuf);
    k_gemm_bf16<1><<<gD, 256, 0, stream>>>(abuf, wo, xbuf, xbuf, D_, D_);       // +res

    k_rmsnorm<<<M_, 256, 0, stream>>>(xbuf, mlp_norm + g * D_, nbuf);
    k_gemm_bf16<0><<<gFF, 256, 0, stream>>>(nbuf, wg, nullptr, gbuf, DFF_, D_); // gate
    k_gemm_bf16<2><<<gFF, 256, 0, stream>>>(nbuf, wu, gbuf, gbuf, DFF_, D_);    // silu(gate)*up
    k_gemm_bf16<1><<<gD,  256, 0, stream>>>(gbuf, wd, xbuf, xbuf, D_, DFF_);    // +res

    for (int c = 0; c < 3; ++c) {
      const int j = 3 * g + c;
      k_rmsnorm<<<M_, 256, 0, stream>>>(xbuf, cnorm + j * D_, nbuf);
      k_zero<<<(B_ * D_ + 64 + 255) / 256, 256, 0, stream>>>(hbuf, B_ * D_ + 64);
      k_cfc_scan<<<CFC_BLOCKS, 256, 0, stream>>>(
          nbuf, Wa1 + (size_t)j * D_ * 2 * D_, Wa2 + (size_t)j * D_ * 2 * D_,
          clogdt + j * D_, hbuf, abuf, ctr);
      k_gemm_bf16<1><<<gD, 256, 0, stream>>>(abuf, bWout + (size_t)j * D_ * D_,
                                             xbuf, xbuf, D_, D_);
    }
  }

  k_rmsnorm<<<M_, 256, 0, stream>>>(xbuf, fnorm, nbuf);
  k_gemm_bf16<0><<<gV, 256, 0, stream>>>(nbuf, bEmb, nullptr, (float*)d_out, V_, D_);
}
